// BilinearBlock_2723009266068
// MI455X (gfx1250) — matrix operand order repaired and checked by reading, not compiled, not run on hardware
//
#include <hip/hip_runtime.h>

typedef __bf16 bf16;
typedef __attribute__((ext_vector_type(16))) __bf16 v16bf;
typedef __attribute__((ext_vector_type(8)))  float  v8f;
typedef __attribute__((ext_vector_type(4)))  unsigned int u32x4;
typedef __attribute__((ext_vector_type(4)))  int v4i;

// Address-space-qualified pointee typedefs (attributes in cast type-ids are
// dropped by clang; typedefs keep them).
typedef __attribute__((address_space(1))) v4i v4i_as1;  // global
typedef __attribute__((address_space(3))) v4i v4i_as3;  // LDS

union FragU { u32x4 q[2]; v16bf v; };

#if __has_builtin(__builtin_amdgcn_global_load_async_to_lds_b128)
#define HAVE_ASYNC_LDS 1
#else
#define HAVE_ASYNC_LDS 0
#endif

// 16-byte global -> LDS copy. Async (ASYNCcnt) when available, else via VGPRs.
__device__ __forceinline__ void cp16(const bf16* g, bf16* l) {
#if HAVE_ASYNC_LDS
  __builtin_amdgcn_global_load_async_to_lds_b128(
      (v4i_as1*)(unsigned long long)g,
      (v4i_as3*)(unsigned int)(unsigned long long)l, 0, 0);
#else
  *(u32x4*)l = *(const u32x4*)g;
#endif
}

template <short N>
__device__ __forceinline__ void wait_async() {
#if HAVE_ASYNC_LDS
#if __has_builtin(__builtin_amdgcn_s_wait_asynccnt)
  __builtin_amdgcn_s_wait_asynccnt(N);
#else
  asm volatile("s_wait_asynccnt %0" ::"i"((int)N) : "memory");
#endif
#endif
}

// ---------------------------------------------------------------------------
// WMMA helper: D = A(16x32 bf16) * B(32x16 bf16) + C(16x16 f32)
// ---------------------------------------------------------------------------
__device__ __forceinline__ v8f wmma_bf16(v16bf a, v16bf b, v8f c) {
  return __builtin_amdgcn_wmma_f32_16x16x32_bf16(false, a, false, b, (short)0, c,
                                                 false, false);
}

// A-fragment (16x32 bf16) from LDS tile stored row-major [rows][32].
// lanes 0-15: K {0..7, 16..23}; lanes 16-31: K {8..15, 24..31}; row = lane&15.
__device__ __forceinline__ v16bf load_a_frag(const bf16* lds, int row_base, int lane) {
  int r  = row_base + (lane & 15);
  int hi = (lane >> 4) & 1;
  const bf16* p = lds + r * 32 + hi * 8;
  FragU u;
  u.q[0] = *(const u32x4*)(p);
  u.q[1] = *(const u32x4*)(p + 16);
  return u.v;
}

// B-fragment (32x16 bf16, B stored transposed in LDS as [cols][32] row-major).
// lane holds column n = lane&15; lanes 0-15: K {0..7, 16..23}; lanes 16-31: K {8..15, 24..31}.
__device__ __forceinline__ v16bf load_b_frag(const bf16* lds, int col_base, int lane) {
  int n  = col_base + (lane & 15);
  int hi = (lane >> 4) & 1;
  const bf16* p = lds + n * 32 + hi * 8;
  FragU u;
  u.q[0] = *(const u32x4*)(p);
  u.q[1] = *(const u32x4*)(p + 16);
  return u.v;
}

// ---------------------------------------------------------------------------
// Generic GEMM:  C[M,N] = A[M,K](bf16,row-major) @ B  where BT[N,K] is given.
// Block tile 128x128, BK=32, 8 waves (2x4), wave tile 64x32 (4x2 WMMA tiles).
// Double-buffered LDS, async global->LDS staging.
// EPI: 0 = store f32, 1 = store bf16, 2 = add residual(f32) and store f32
// ---------------------------------------------------------------------------
template <int EPI>
__global__ __launch_bounds__(256) void gemm_bt_kernel(
    const bf16* __restrict__ A, const bf16* __restrict__ BT,
    const float* __restrict__ Res, void* __restrict__ Out,
    int M, int N, int K,
    long long strideA, long long strideB, long long strideO) {
  __shared__ bf16 sA[2][128 * 32];
  __shared__ bf16 sB[2][128 * 32];

  A  += (long long)blockIdx.z * strideA;
  BT += (long long)blockIdx.z * strideB;

  const int bm   = blockIdx.x * 128;
  const int bn   = blockIdx.y * 128;
  const int tid  = threadIdx.x;
  const int lane = tid & 31;
  const int wave = tid >> 5;
  const int wm   = wave >> 2;   // 0..1 -> 64 rows each
  const int wn   = wave & 3;    // 0..3 -> 32 cols each

  v8f acc[4][2] = {};

  const int row2 = tid >> 1;        // 0..127 (2 threads per tile row)
  const int e0   = (tid & 1) * 16;  // element offset within 32-wide row

  const bf16* gA = A + (size_t)(bm + row2) * K + e0;
  const bf16* gB = BT + (size_t)(bn + row2) * K + e0;

  auto issue = [&](int k0, int buf) {
    bf16* la = &sA[buf][row2 * 32 + e0];
    bf16* lb = &sB[buf][row2 * 32 + e0];
    cp16(gA + k0, la);
    cp16(gA + k0 + 8, la + 8);
    cp16(gB + k0, lb);
    cp16(gB + k0 + 8, lb + 8);
  };

  issue(0, 0);
  const int T = K >> 5;
  for (int t = 0; t < T; ++t) {
    const int cur = t & 1;
    if (t + 1 < T) {
      issue((t + 1) << 5, cur ^ 1);
      wait_async<4>();  // 4 in flight for next tile; current tile complete
    } else {
      wait_async<0>();
    }
    __syncthreads();

    v16bf af[4], bf[2];
#pragma unroll
    for (int i = 0; i < 4; ++i) af[i] = load_a_frag(sA[cur], wm * 64 + i * 16, lane);
#pragma unroll
    for (int j = 0; j < 2; ++j) bf[j] = load_b_frag(sB[cur], wn * 32 + j * 16, lane);
#pragma unroll
    for (int i = 0; i < 4; ++i)
#pragma unroll
      for (int j = 0; j < 2; ++j) acc[i][j] = wmma_bf16(af[i], bf[j], acc[i][j]);
    __syncthreads();  // all reads of buf[cur] done before it is refilled
  }

  // Epilogue: C 16x16 f32 layout: vgpr v -> row v + 8*(lane>=16), col = lane&15
  const int ncol  = bn + wn * 32 + (lane & 15);
  const int mbase = (lane >> 4) * 8;
  float* of = (float*)Out + (long long)blockIdx.z * strideO;
  bf16*  ob = (bf16*)Out + (long long)blockIdx.z * strideO;
#pragma unroll
  for (int i = 0; i < 4; ++i)
#pragma unroll
    for (int j = 0; j < 2; ++j) {
      int r0 = bm + wm * 64 + i * 16 + mbase;
      int cc = ncol + j * 16;
#pragma unroll
      for (int v = 0; v < 8; ++v) {
        size_t idx = (size_t)(r0 + v) * N + cc;
        float  val = acc[i][j][v];
        if constexpr (EPI == 0)      of[idx] = val;
        else if constexpr (EPI == 1) ob[idx] = (bf16)val;
        else                         of[idx] = val + Res[idx];
      }
    }
}

// ---------------------------------------------------------------------------
// Dual GEMM:  Out = epi( (A1@B1) * (A2@B2) )  elementwise, bf16 out.
// Block tile 128x64, 8 waves (4x2), wave tile 32x32 (2x2 tiles per product).
// Double-buffered LDS, async staging.
// MASKED: causal mask (col>row -> 0); scale2 applied always.
// ---------------------------------------------------------------------------
template <bool MASKED>
__global__ __launch_bounds__(256) void dual_gemm_kernel(
    const bf16* __restrict__ A1, const bf16* __restrict__ A2,
    const bf16* __restrict__ B1T, const bf16* __restrict__ B2T,
    bf16* __restrict__ Out, int M, int N, int K, float scale2,
    long long strideA, long long strideB, long long strideO) {
  const int bm = blockIdx.x * 128;
  const int bn = blockIdx.y * 64;
  const int tid = threadIdx.x;
  Out += (long long)blockIdx.z * strideO;

  if (MASKED && bn > bm + 127) {  // tile fully above the diagonal -> zeros
    for (int i = tid; i < 128 * 64; i += 256) {
      int r = i >> 6, c = i & 63;
      Out[(size_t)(bm + r) * N + (bn + c)] = (bf16)0.0f;
    }
    return;
  }

  __shared__ bf16 sA1[2][128 * 32];
  __shared__ bf16 sA2[2][128 * 32];
  __shared__ bf16 sB1[2][64 * 32];
  __shared__ bf16 sB2[2][64 * 32];

  A1  += (long long)blockIdx.z * strideA;
  A2  += (long long)blockIdx.z * strideA;
  B1T += (long long)blockIdx.z * strideB;
  B2T += (long long)blockIdx.z * strideB;

  const int lane = tid & 31;
  const int wave = tid >> 5;
  const int wm   = wave >> 1;  // 0..3 -> 32 rows each
  const int wn   = wave & 1;   // 0..1 -> 32 cols each

  v8f acc1[2][2] = {}, acc2[2][2] = {};

  const int rowA = tid >> 1;        // 0..127, 2 threads/row
  const int eA   = (tid & 1) * 16;
  const int rowB = tid >> 2;        // 0..63, 4 threads/row
  const int eB   = (tid & 3) * 8;

  const bf16* g1 = A1 + (size_t)(bm + rowA) * K + eA;
  const bf16* g2 = A2 + (size_t)(bm + rowA) * K + eA;
  const bf16* h1 = B1T + (size_t)(bn + rowB) * K + eB;
  const bf16* h2 = B2T + (size_t)(bn + rowB) * K + eB;

  auto issue = [&](int k0, int buf) {
    bf16* la1 = &sA1[buf][rowA * 32 + eA];
    bf16* la2 = &sA2[buf][rowA * 32 + eA];
    cp16(g1 + k0, la1);
    cp16(g1 + k0 + 8, la1 + 8);
    cp16(g2 + k0, la2);
    cp16(g2 + k0 + 8, la2 + 8);
    cp16(h1 + k0, &sB1[buf][rowB * 32 + eB]);
    cp16(h2 + k0, &sB2[buf][rowB * 32 + eB]);
  };

  issue(0, 0);
  const int T = K >> 5;
  for (int t = 0; t < T; ++t) {
    const int cur = t & 1;
    if (t + 1 < T) {
      issue((t + 1) << 5, cur ^ 1);
      wait_async<6>();
    } else {
      wait_async<0>();
    }
    __syncthreads();

    v16bf af1[2], af2[2], bf1[2], bf2[2];
#pragma unroll
    for (int i = 0; i < 2; ++i) {
      af1[i] = load_a_frag(sA1[cur], wm * 32 + i * 16, lane);
      af2[i] = load_a_frag(sA2[cur], wm * 32 + i * 16, lane);
      bf1[i] = load_b_frag(sB1[cur], wn * 32 + i * 16, lane);
      bf2[i] = load_b_frag(sB2[cur], wn * 32 + i * 16, lane);
    }
#pragma unroll
    for (int i = 0; i < 2; ++i)
#pragma unroll
      for (int j = 0; j < 2; ++j) {
        acc1[i][j] = wmma_bf16(af1[i], bf1[j], acc1[i][j]);
        acc2[i][j] = wmma_bf16(af2[i], bf2[j], acc2[i][j]);
      }
    __syncthreads();
  }

  const int ncol  = bn + wn * 32 + (lane & 15);
  const int mbase = (lane >> 4) * 8;
#pragma unroll
  for (int i = 0; i < 2; ++i)
#pragma unroll
    for (int j = 0; j < 2; ++j) {
      int r0 = bm + wm * 32 + i * 16 + mbase;
      int cc = ncol + j * 16;
#pragma unroll
      for (int v = 0; v < 8; ++v) {
        int   r   = r0 + v;
        float val = acc1[i][j][v] * acc2[i][j][v] * scale2;
        if (MASKED && cc > r) val = 0.0f;
        Out[(size_t)r * N + cc] = (bf16)val;
      }
    }
}

// ---------------------------------------------------------------------------
// RMSNorm over last dim D (=1024) + cast to bf16.  One block per row.
// ---------------------------------------------------------------------------
__global__ __launch_bounds__(256) void rmsnorm_kernel(
    const float* __restrict__ X, bf16* __restrict__ Out, int D) {
  const size_t row = blockIdx.x;
  const float* x = X + row * D;
  float vals[4];
  float s = 0.0f;
#pragma unroll
  for (int i = 0; i < 4; ++i) {
    vals[i] = x[threadIdx.x + i * 256];
    s += vals[i] * vals[i];
  }
  __shared__ float red[256];
  red[threadIdx.x] = s;
  __syncthreads();
  for (int off = 128; off > 0; off >>= 1) {
    if (threadIdx.x < off) red[threadIdx.x] += red[threadIdx.x + off];
    __syncthreads();
  }
  float inv = rsqrtf(red[0] / (float)D + 1e-6f);
#pragma unroll
  for (int i = 0; i < 4; ++i)
    Out[row * D + threadIdx.x + i * 256] = (bf16)(vals[i] * inv);
}

// ---------------------------------------------------------------------------
// RoPE + cast: In f32 [rows,128] -> Out bf16 [rows,128]; pos = row % S.
// ---------------------------------------------------------------------------
__global__ void rope_cast_kernel(const float* __restrict__ In,
                                 const float* __restrict__ Cos,
                                 const float* __restrict__ Sin,
                                 bf16* __restrict__ Out, int S) {
  int idx = blockIdx.x * blockDim.x + threadIdx.x;  // rows*64 threads
  int row = idx >> 6, j = idx & 63;
  int s   = row & (S - 1);
  float x1 = In[(size_t)row * 128 + j];
  float x2 = In[(size_t)row * 128 + 64 + j];
  float c  = Cos[(size_t)s * 64 + j];
  float sn = Sin[(size_t)s * 64 + j];
  Out[(size_t)row * 128 + j]      = (bf16)(x1 * c - x2 * sn);
  Out[(size_t)row * 128 + 64 + j] = (bf16)(x1 * sn + x2 * c);
}

// v f32 [Bt][S][128] -> vT bf16 [Bt][128][S]
__global__ void transpose_cast_v(const float* __restrict__ In,
                                 bf16* __restrict__ Out) {
  int idx = blockIdx.x * blockDim.x + threadIdx.x;  // Bt*S*128
  int d = idx & 127, s = (idx >> 7) & 2047, b = idx >> 18;
  Out[((size_t)b * 128 + d) * 2048 + s] = (bf16)In[idx];
}

// W f32 [din][dout] -> WT bf16 [dout][din]
__global__ void convt_kernel(const float* __restrict__ In, bf16* __restrict__ Out,
                             int din, int dout) {
  int idx = blockIdx.x * blockDim.x + threadIdx.x;
  if (idx >= din * dout) return;
  int n = idx % dout, k = idx / dout;
  Out[(size_t)n * din + k] = (bf16)In[idx];
}

// ---------------------------------------------------------------------------
// Host orchestration
// ---------------------------------------------------------------------------
static inline char* carve(char*& p, size_t bytes) {
  char* r = p;
  p += (bytes + 255) & ~(size_t)255;
  return r;
}

extern "C" void kernel_launch(void* const* d_in, const int* in_sizes, int n_in,
                              void* d_out, int out_size, void* d_ws, size_t ws_size,
                              hipStream_t stream) {
  constexpr int Bt = 4, S = 2048, DM = 1024, DH = 128, DMLP = 4096;
  constexpr int R = Bt * S;  // 8192 rows

  const float* x    = (const float*)d_in[0];
  const float* cosb = (const float*)d_in[1];
  const float* sinb = (const float*)d_in[2];
  const float* wq1  = (const float*)d_in[4];
  const float* wq2  = (const float*)d_in[5];
  const float* wk1  = (const float*)d_in[6];
  const float* wk2  = (const float*)d_in[7];
  const float* wv   = (const float*)d_in[8];
  const float* wo   = (const float*)d_in[9];
  const float* wm   = (const float*)d_in[10];
  const float* wn   = (const float*)d_in[11];
  const float* wp   = (const float*)d_in[12];

  char* p = (char*)d_ws;
  bf16* wq1T = (bf16*)carve(p, (size_t)DH * DM * 2);
  bf16* wq2T = (bf16*)carve(p, (size_t)DH * DM * 2);
  bf16* wk1T = (bf16*)carve(p, (size_t)DH * DM * 2);
  bf16* wk2T = (bf16*)carve(p, (size_t)DH * DM * 2);
  bf16* wvT  = (bf16*)carve(p, (size_t)DH * DM * 2);
  bf16* woT  = (bf16*)carve(p, (size_t)DM * DH * 2);
  bf16* wmT  = (bf16*)carve(p, (size_t)DMLP * DM * 2);
  bf16* wnT  = (bf16*)carve(p, (size_t)DMLP * DM * 2);
  bf16* wpT  = (bf16*)carve(p, (size_t)DM * DMLP * 2);
  bf16* xn   = (bf16*)carve(p, (size_t)R * DM * 2);
  float* q1r = (float*)carve(p, (size_t)R * DH * 4);
  float* q2r = (float*)carve(p, (size_t)R * DH * 4);
  float* k1r = (float*)carve(p, (size_t)R * DH * 4);
  float* k2r = (float*)carve(p, (size_t)R * DH * 4);
  float* vr  = (float*)carve(p, (size_t)R * DH * 4);
  bf16* q1b  = (bf16*)carve(p, (size_t)R * DH * 2);
  bf16* q2b  = (bf16*)carve(p, (size_t)R * DH * 2);
  bf16* k1b  = (bf16*)carve(p, (size_t)R * DH * 2);
  bf16* k2b  = (bf16*)carve(p, (size_t)R * DH * 2);
  bf16* vT   = (bf16*)carve(p, (size_t)Bt * DH * S * 2);
  bf16* attn = (bf16*)carve(p, (size_t)Bt * S * S * 2);
  bf16* ao   = (bf16*)carve(p, (size_t)R * DH * 2);
  float* x2  = (float*)carve(p, (size_t)R * DM * 4);
  bf16* xn2  = (bf16*)carve(p, (size_t)R * DM * 2);
  bf16* hg   = (bf16*)carve(p, (size_t)R * DMLP * 2);

  auto cvt = [&](const float* w, bf16* wt, int din, int dout) {
    int n = din * dout;
    convt_kernel<<<(n + 255) / 256, 256, 0, stream>>>(w, wt, din, dout);
  };
  cvt(wq1, wq1T, DM, DH);
  cvt(wq2, wq2T, DM, DH);
  cvt(wk1, wk1T, DM, DH);
  cvt(wk2, wk2T, DM, DH);
  cvt(wv, wvT, DM, DH);
  cvt(wo, woT, DH, DM);
  cvt(wm, wmT, DM, DMLP);
  cvt(wn, wnT, DM, DMLP);
  cvt(wp, wpT, DMLP, DM);

  // xn = rmsnorm(x)
  rmsnorm_kernel<<<R, 256, 0, stream>>>(x, xn, DM);

  // projections: [R,DM] @ [DM,DH]
  dim3 gP(R / 128, DH / 128, 1);
  gemm_bt_kernel<0><<<gP, 256, 0, stream>>>(xn, wq1T, nullptr, q1r, R, DH, DM, 0, 0, 0);
  gemm_bt_kernel<0><<<gP, 256, 0, stream>>>(xn, wq2T, nullptr, q2r, R, DH, DM, 0, 0, 0);
  gemm_bt_kernel<0><<<gP, 256, 0, stream>>>(xn, wk1T, nullptr, k1r, R, DH, DM, 0, 0, 0);
  gemm_bt_kernel<0><<<gP, 256, 0, stream>>>(xn, wk2T, nullptr, k2r, R, DH, DM, 0, 0, 0);
  gemm_bt_kernel<0><<<gP, 256, 0, stream>>>(xn, wvT, nullptr, vr, R, DH, DM, 0, 0, 0);

  // rope + casts
  int ropeN = R * 64;
  rope_cast_kernel<<<ropeN / 256, 256, 0, stream>>>(q1r, cosb, sinb, q1b, S);
  rope_cast_kernel<<<ropeN / 256, 256, 0, stream>>>(q2r, cosb, sinb, q2b, S);
  rope_cast_kernel<<<ropeN / 256, 256, 0, stream>>>(k1r, cosb, sinb, k1b, S);
  rope_cast_kernel<<<ropeN / 256, 256, 0, stream>>>(k2r, cosb, sinb, k2b, S);
  transpose_cast_v<<<(R * DH) / 256, 256, 0, stream>>>(vr, vT);

  // bilinear masked scores: a = mask((q1k1^T)(q2k2^T)/DH), per batch
  dim3 gS(S / 128, S / 64, Bt);
  dual_gemm_kernel<true><<<gS, 256, 0, stream>>>(
      q1b, q2b, k1b, k2b, attn, S, S, DH, 1.0f / (float)DH,
      (long long)S * DH, (long long)S * DH, (long long)S * S);

  // attn_out = a @ v  (bf16 result)
  dim3 gAV(S / 128, DH / 128, Bt);
  gemm_bt_kernel<1><<<gAV, 256, 0, stream>>>(
      attn, vT, nullptr, ao, S, DH, S,
      (long long)S * S, (long long)DH * S, (long long)S * DH);

  // x2 = x + attn_out @ wo
  dim3 gO(R / 128, DM / 128, 1);
  gemm_bt_kernel<2><<<gO, 256, 0, stream>>>(ao, woT, x, x2, R, DM, DH, 0, 0, 0);

  // xn2 = rmsnorm(x2)
  rmsnorm_kernel<<<R, 256, 0, stream>>>(x2, xn2, DM);

  // gated MLP hidden: hg = (xn2@wm) * (xn2@wn)
  dim3 gM(R / 128, DMLP / 64, 1);
  dual_gemm_kernel<false><<<gM, 256, 0, stream>>>(
      xn2, xn2, wmT, wnT, hg, R, DMLP, DM, 1.0f, 0, 0, 0);

  // out = x2 + hg @ wp
  gemm_bt_kernel<2><<<gO, 256, 0, stream>>>(hg, wpT, x2, (float*)d_out, R, DM, DMLP,
                                            0, 0, 0);
}
